// ExchangeForecaster_64175401337576
// MI455X (gfx1250) — matrix operand order repaired and checked by reading, not compiled, not run on hardware
//
#include <hip/hip_runtime.h>
#include <hip/hip_bf16.h>

#define CTX   336
#define PRED  30
#define NSAMP 100
#define HID   128
#define BATCH 1024
#define TPAST 512
#define K1    4832          // [th(2) + static(4824)] padded 4826 -> 4832 (mult of 32)
#define KSTEPS (K1/32)      // 151
#define ZTOT  (NSAMP*BATCH*PRED)   // 3,072,000
#define ZHALF (ZTOT/2)             // 1,536,000

typedef __attribute__((ext_vector_type(16))) _Float16 v16h;
typedef __attribute__((ext_vector_type(8)))  float    v8f;
typedef __attribute__((ext_vector_type(4)))  unsigned v4u;
typedef __attribute__((ext_vector_type(8)))  int      v8i;
typedef __attribute__((ext_vector_type(4)))  int      v4i;

#define WMMA_F16(a, b, c) \
  __builtin_amdgcn_wmma_f32_16x16x32_f16(false, (a), false, (b), (short)0, (c), false, false)

// ---------------------------------------------------------------------------
// Tensor Data Mover: 2D tile load Global->LDS (D# per CDNA5 ISA ch.8).
// data_size = 2 bytes (code 1). dims/strides in elements. Tile assumed in-bounds.
// clang-23 toolchain: 6-arg builtin (g0 v4u, g1 v8i, g2 v4i, g3 v4i, g4 v8i, cpol)
__device__ __forceinline__ void tdm_load_2d(void* lds_dst, const void* gsrc,
                                            unsigned tile_x, unsigned tile_y,
                                            unsigned tensor_x, unsigned tensor_y,
                                            unsigned stride_x) {
  const unsigned long long ga = (unsigned long long)(uintptr_t)gsrc;
  const unsigned lds = (unsigned)(uintptr_t)lds_dst;   // low 32b of LDS aperture ptr
  v4u g0;
  g0[0] = 1u;                                              // count=1 (valid user D#)
  g0[1] = lds;                                             // lds_addr [63:32]
  g0[2] = (unsigned)ga;                                    // global_addr lo [95:64]
  g0[3] = (unsigned)((ga >> 32) & 0x1FFFFFFu) | (2u << 30);// addr[56:32] | type=2
  v8i g1;
  g1[0] = 0x00010000;                                      // data_size=1 (2B), mask=0
  g1[1] = (int)((tensor_x & 0xFFFFu) << 16);               // tensor_dim0[15:0] @ bit48
  g1[2] = (int)(((tensor_x >> 16) & 0xFFFFu) |
                ((tensor_y & 0xFFFFu) << 16));             // dim0 hi | dim1 lo
  g1[3] = (int)(((tensor_y >> 16) & 0xFFFFu) |
                ((tile_x & 0xFFFFu) << 16));               // dim1 hi | tile_dim0
  g1[4] = (int)(tile_y & 0xFFFFu);                         // tile_dim1 (tile_dim2=0)
  g1[5] = (int)stride_x;                                   // tensor_dim0_stride lo32
  g1[6] = 0;                                               // stride hi16 | dim1_stride lo
  g1[7] = 0;
  v4i g2 = {0, 0, 0, 0}, g3 = {0, 0, 0, 0};                // 2D: groups 2/3 unused
  v8i g4 = {0, 0, 0, 0, 0, 0, 0, 0};                       // extended group (unused)
  __builtin_amdgcn_tensor_load_to_lds(g0, g1, g2, g3, g4, 0);
}

// ---------------------------------------------------------------------------
// WMMA fragment loaders (ISA 7.12.2 layouts, wave32)
__device__ __forceinline__ v16h load_a_frag(const _Float16* __restrict__ src,
                                            int ld, int m0, int k0, int lane) {
  const int m = m0 + (lane & 15);
  const int half = lane >> 4;
  const _Float16* r = src + (size_t)m * ld + k0;
  v16h a;
#pragma unroll
  for (int v = 0; v < 8; ++v) {
    const int kb = ((v & 4) << 2) + half * 8 + ((v & 3) << 1);
    a[2 * v]     = r[kb];
    a[2 * v + 1] = r[kb + 1];
  }
  return a;
}
__device__ __forceinline__ v16h load_b_frag(const _Float16* __restrict__ src,
                                            int ld, int k0, int n0, int lane) {
  const _Float16* r = src + (size_t)k0 * ld + n0 + (lane & 15);
  const int half = lane >> 4;
  v16h b;
#pragma unroll
  for (int i = 0; i < 16; ++i) b[i] = r[(size_t)(((i & 8) << 1) + half * 8 + (i & 7)) * ld];
  return b;
}
// D f32 16x16: lane: n = lane&15, mhalf = lane>>4; VGPR r -> m = r + 8*mhalf.

// ---------------------------------------------------------------------------
// JAX threefry2x32 with key(42) = {0, 42}; normal via sqrt(2)*erfinv(uniform(-1,1))
__device__ __forceinline__ unsigned rotl32(unsigned x, int r) {
  return (x << r) | (x >> (32 - r));
}
__device__ __forceinline__ void threefry2x32(unsigned c0, unsigned c1,
                                             unsigned& o0, unsigned& o1) {
  const unsigned k0 = 0u, k1 = 42u, k2 = 0x1BD11BDAu ^ k0 ^ k1;
  unsigned x0 = c0 + k0, x1 = c1 + k1;
#pragma unroll
  for (int i = 0; i < 4; ++i) { const int R[4] = {13,15,26,6};  x0 += x1; x1 = rotl32(x1, R[i]); x1 ^= x0; }
  x0 += k1; x1 += k2 + 1u;
#pragma unroll
  for (int i = 0; i < 4; ++i) { const int R[4] = {17,29,16,24}; x0 += x1; x1 = rotl32(x1, R[i]); x1 ^= x0; }
  x0 += k2; x1 += k0 + 2u;
#pragma unroll
  for (int i = 0; i < 4; ++i) { const int R[4] = {13,15,26,6};  x0 += x1; x1 = rotl32(x1, R[i]); x1 ^= x0; }
  x0 += k0; x1 += k1 + 3u;
#pragma unroll
  for (int i = 0; i < 4; ++i) { const int R[4] = {17,29,16,24}; x0 += x1; x1 = rotl32(x1, R[i]); x1 ^= x0; }
  x0 += k1; x1 += k2 + 4u;
#pragma unroll
  for (int i = 0; i < 4; ++i) { const int R[4] = {13,15,26,6};  x0 += x1; x1 = rotl32(x1, R[i]); x1 ^= x0; }
  x0 += k2; x1 += k0 + 5u;
  o0 = x0; o1 = x1;
}
__device__ __forceinline__ float bits_to_float01(unsigned bits) {
  union { unsigned u; float f; } cvt;
  cvt.u = (bits >> 9) | 0x3f800000u;   // [1,2)
  return cvt.f - 1.0f;                 // [0,1)
}
__device__ __forceinline__ float erfinv_f(float x) {
  float w = -logf((1.0f - x) * (1.0f + x));
  float p;
  if (w < 5.0f) {
    w -= 2.5f;
    p = 2.81022636e-08f;
    p = fmaf(p, w, 3.43273939e-07f);
    p = fmaf(p, w, -3.5233877e-06f);
    p = fmaf(p, w, -4.39150654e-06f);
    p = fmaf(p, w, 0.00021858087f);
    p = fmaf(p, w, -0.00125372503f);
    p = fmaf(p, w, -0.00417768164f);
    p = fmaf(p, w, 0.246640727f);
    p = fmaf(p, w, 1.50140941f);
  } else {
    w = sqrtf(w) - 3.0f;
    p = -0.000200214257f;
    p = fmaf(p, w, 0.000100950558f);
    p = fmaf(p, w, 0.00134934322f);
    p = fmaf(p, w, -0.00367342844f);
    p = fmaf(p, w, 0.00573950773f);
    p = fmaf(p, w, -0.0076224613f);
    p = fmaf(p, w, 0.00943887047f);
    p = fmaf(p, w, 1.00167406f);
    p = fmaf(p, w, 2.83297682f);
  }
  return p * x;
}
__device__ __forceinline__ float z_normal(unsigned i) {
  unsigned o0, o1, bits;
  if (i < (unsigned)ZHALF) { threefry2x32(i, i + ZHALF, o0, o1); bits = o0; }
  else                     { threefry2x32(i - ZHALF, i, o0, o1); bits = o1; }
  const float lo = -0.99999994f, hi = 1.0f;
  float u = lo + bits_to_float01(bits) * (hi - lo);
  u = fmaxf(u, lo);
  return 1.4142135623730951f * erfinv_f(u);
}

__device__ __forceinline__ float gelu_f(float x) {
  const float c = 0.7978845608028654f;  // sqrt(2/pi)
  return 0.5f * x * (1.0f + tanhf(c * (x + 0.044715f * x * x * x)));
}

__device__ __forceinline__ float block_sum(float v, float* s) {
  const int t = threadIdx.x;
  s[t] = v; __syncthreads();
#pragma unroll
  for (int o = 128; o > 0; o >>= 1) { if (t < o) s[t] += s[t + o]; __syncthreads(); }
  const float r = s[0]; __syncthreads();
  return r;
}

// ---------------------------------------------------------------------------
// Kernel 1: weight repack f32 -> f16 (split + pad)
__global__ void __launch_bounds__(256)
fc_convert(const float* __restrict__ W1, const float* __restrict__ W2,
           const float* __restrict__ W3, _Float16* __restrict__ W1s,
           _Float16* __restrict__ W1z, _Float16* __restrict__ W2h,
           _Float16* __restrict__ W3h) {
  int i = blockIdx.x * 256 + threadIdx.x;
  const int N1 = K1 * HID;                       // 618496
  if (i < N1) { int k = i >> 7, n = i & 127;
    W1s[i] = (_Float16)((k < 4826) ? W1[(size_t)(k + 30) * HID + n] : 0.0f); return; }
  i -= N1;
  if (i < 32 * HID) { int k = i >> 7, n = i & 127;
    W1z[i] = (_Float16)((k < PRED) ? W1[(size_t)k * HID + n] : 0.0f); return; }
  i -= 32 * HID;
  if (i < HID * HID) { W2h[i] = (_Float16)W2[i]; return; }
  i -= HID * HID;
  if (i < HID * 32) { int k = i >> 5, n = i & 31;
    W3h[i] = (_Float16)((n < PRED) ? W3[(size_t)k * PRED + n] : 0.0f); return; }
}

// ---------------------------------------------------------------------------
// Kernel 2: per-row feature build -> f16 Xs row (K1 wide), loc/scale
__global__ void __launch_bounds__(256)
fc_features(const float* __restrict__ pt, const float* __restrict__ ptf,
            const float* __restrict__ ftf, _Float16* __restrict__ Xs,
            float* __restrict__ locp, float* __restrict__ scalep) {
  __shared__ float s_ctx[344];     // pt[T-343 .. T-1]; ctx[t] = s_ctx[7+t]
  __shared__ float s_sort[512];
  __shared__ float s_red[256];
  __shared__ float s_diff[CTX];
  __shared__ float s_rstd[CTX];
  const int b = blockIdx.x, t = threadIdx.x;

  for (int i = t; i < 343; i += 256) s_ctx[i] = pt[(size_t)b * TPAST + 169 + i];
  __syncthreads();
  for (int i = t; i < 512; i += 256) s_sort[i] = (i < CTX) ? s_ctx[7 + i] : 3.0e38f;
  __syncthreads();
  // bitonic sort 512 (ascending)
  for (int k = 2; k <= 512; k <<= 1)
    for (int j = k >> 1; j > 0; j >>= 1) {
      for (int i = t; i < 512; i += 256) {
        const int l = i ^ j;
        if (l > i) {
          const float a = s_sort[i], c = s_sort[l];
          const bool up = ((i & k) == 0);
          if ((a > c) == up) { s_sort[i] = c; s_sort[l] = a; }
        }
      }
      __syncthreads();
    }
  const float loc = 0.5f * (s_sort[167] + s_sort[168]);   // median of 336

  float ps = 0.0f;
  for (int i = t; i < CTX; i += 256) ps += fabsf(s_ctx[7 + i] - loc);
  const float scale = fmaxf(block_sum(ps, s_red) * (1.0f / CTX), 1e-6f);

  float pd = 0.0f;
  for (int i = t; i < CTX; i += 256) {
    const float d = (i == 0) ? 0.0f : (s_ctx[7 + i] - s_ctx[6 + i]);
    s_diff[i] = d; pd += fabsf(d);
  }
  const float dn = fmaxf(block_sum(pd, s_red) * (1.0f / CTX), 1e-6f);

  float pr = 0.0f;
  for (int i = t; i < CTX; i += 256) {
    const int a = (i > 4) ? (i - 4) : 0;
    const float n = (float)(i + 1 - a);
    float s = 0.0f, s2 = 0.0f;
    for (int q = a; q <= i; ++q) { const float v = s_ctx[7 + q]; s += v; s2 += v * v; }
    const float var = (s2 - s * s / n) / fmaxf(n - 1.0f, 1.0f);
    const float r = (i >= 2) ? sqrtf(fmaxf(var, 1e-12f)) : 0.0f;
    s_rstd[i] = r; pr += fabsf(r);
  }
  const float rn = fmaxf(block_sum(pr, s_red) * (1.0f / CTX), 1e-6f);

  _Float16* row = Xs + (size_t)b * K1;
  if (t == 0) { row[0] = (_Float16)1.0f; row[1] = (_Float16)1.0f;
                locp[b] = loc; scalep[b] = scale; }
  if (t < 6) row[4826 + t] = (_Float16)0.0f;
  const float inv = 1.0f / scale;
  for (int e = t; e < 8 * CTX; e += 256) {            // 8 lag channels, normalized
    const int c = e / CTX, tt = e - c * CTX;
    row[2 + e] = (_Float16)((s_ctx[7 - c + tt] - loc) * inv);
  }
  for (int tt = t; tt < CTX; tt += 256) {             // diff, rstd channels
    row[2 + 8 * CTX + tt] = (_Float16)(s_diff[tt] / dn);
    row[2 + 9 * CTX + tt] = (_Float16)(s_rstd[tt] / rn);
  }
  for (int e = t; e < 4 * CTX; e += 256) {            // past time features
    const int f = e / CTX, tt = e - f * CTX;
    row[2 + 10 * CTX + e] = (_Float16)ptf[(size_t)b * TPAST * 4 + (176 + tt) * 4 + f];
  }
  for (int e = t; e < 4 * PRED; e += 256) {           // future time features
    const int f = e / PRED, p = e - f * PRED;
    row[2 + 14 * CTX + e] = (_Float16)ftf[(size_t)b * PRED * 4 + p * 4 + f];
  }
}

// ---------------------------------------------------------------------------
// Kernel 3: base = [th,static] @ W1[30:,:] + b1  (1024x128x4832 f16 WMMA GEMM)
// TDM double-buffered tile pipeline: wave0 issues tensor_load_to_lds for step
// kt+1 while all waves run WMMA on step kt; completion via s_wait_tensorcnt.
__global__ void __launch_bounds__(256)
fc_base_gemm(const _Float16* __restrict__ Xs, const _Float16* __restrict__ W1s,
             const float* __restrict__ b1, float* __restrict__ basep) {
  __shared__ _Float16 At[2][128 * 32];   // [m][k]
  __shared__ _Float16 Bt[2][32 * 128];   // [k][n]
  const int tid = threadIdx.x, w = tid >> 5, lane = tid & 31;
  const int rowBase = blockIdx.x * 128;
  v8f acc[8] = {};

  if (w == 0) {  // prologue: DMA tiles for kt=0
    tdm_load_2d(At[0], Xs + (size_t)rowBase * K1, 32, 128, K1, BATCH, K1);
    tdm_load_2d(Bt[0], W1s, 128, 32, HID, K1, HID);
  }
  for (int kt = 0; kt < KSTEPS; ++kt) {
    if (w == 0) {
      if (kt + 1 < KSTEPS) {  // DMA next tiles into the other buffer
        const int k0n = (kt + 1) * 32;
        tdm_load_2d(At[(kt + 1) & 1], Xs + (size_t)rowBase * K1 + k0n,
                    32, 128, K1, BATCH, K1);
        tdm_load_2d(Bt[(kt + 1) & 1], W1s + (size_t)k0n * HID,
                    128, 32, HID, K1, HID);
        __builtin_amdgcn_s_wait_tensorcnt((short)2);  // pair kt landed
      } else {
        __builtin_amdgcn_s_wait_tensorcnt((short)0);
      }
    }
    __syncthreads();
    const _Float16* Ab = At[kt & 1];
    const _Float16* Bb = Bt[kt & 1];
    const v16h a = load_a_frag(Ab, 32, w * 16, 0, lane);
#pragma unroll
    for (int nt = 0; nt < 8; ++nt) {
      const v16h bf = load_b_frag(Bb, 128, 0, nt * 16, lane);
      acc[nt] = WMMA_F16(a, bf, acc[nt]);
    }
    __syncthreads();  // all reads done before buffer kt&1 is overwritten
  }
  const int mh = lane >> 4, nn = lane & 15;
#pragma unroll
  for (int nt = 0; nt < 8; ++nt) {
    const int n = nt * 16 + nn;
    const float bb = b1[n];
#pragma unroll
    for (int r = 0; r < 8; ++r) {
      const int m = w * 16 + r + 8 * mh;
      basep[(size_t)(rowBase + m) * HID + n] = acc[nt][r] + bb;
    }
  }
}

// ---------------------------------------------------------------------------
// Kernel 4: per-sample fused MLP. grid = (8 row-tiles, 100 samples), 256 thr.
__global__ void __launch_bounds__(256)
fc_samples(const _Float16* __restrict__ W1z, const _Float16* __restrict__ W2h,
           const _Float16* __restrict__ W3h, const float* __restrict__ basep,
           const float* __restrict__ b2, const float* __restrict__ b3,
           const float* __restrict__ locp, const float* __restrict__ scalep,
           float* __restrict__ out) {
  __shared__ _Float16 zh[128 * 32];      // z tile (f16, padded to 32)
  __shared__ _Float16 w1z[32 * 128];     // W1 z-part (TDM-loaded)
  __shared__ _Float16 hbuf[128 * 128];   // h1 then h2 (wave-private 16-row stripes)
  const int tid = threadIdx.x, w = tid >> 5, lane = tid & 31;
  const int rowBase = blockIdx.x * 128, s = blockIdx.y;
  const int mh = lane >> 4, nn = lane & 15;

  if (w == 0)  // DMA W1z while everyone runs threefry below (overlap)
    tdm_load_2d(w1z, W1z, 4096, 1, 4096, 1, 4096);
  __builtin_prefetch(W2h + tid * 64, 0, 1);   // warm W2/W3 into caches
  __builtin_prefetch(W3h + tid * 16, 0, 1);

  for (int e = tid; e < 128 * PRED; e += 256) {     // exact JAX threefry z
    const int r = e / PRED, p = e - r * PRED;
    const unsigned idx = (unsigned)(s * (BATCH * PRED) + (rowBase + r) * PRED + p);
    zh[r * 32 + p] = (_Float16)z_normal(idx);
  }
  { const int r = tid >> 1, c = 30 + (tid & 1); zh[r * 32 + c] = (_Float16)0.0f; }
  if (w == 0) __builtin_amdgcn_s_wait_tensorcnt((short)0);
  __syncthreads();

  // stage 1: h1 = gelu(z @ W1z + base)           (K=32, one WMMA per tile)
  {
    v8f acc[8] = {};
    const v16h a = load_a_frag(zh, 32, w * 16, 0, lane);
#pragma unroll
    for (int nt = 0; nt < 8; ++nt) {
      const v16h bf = load_b_frag(w1z, 128, 0, nt * 16, lane);
      acc[nt] = WMMA_F16(a, bf, acc[nt]);
    }
#pragma unroll
    for (int nt = 0; nt < 8; ++nt) {
      const int n = nt * 16 + nn;
#pragma unroll
      for (int r = 0; r < 8; ++r) {
        const int m = w * 16 + r + 8 * mh;
        const float v = acc[nt][r] + basep[(size_t)(rowBase + m) * HID + n];
        hbuf[m * HID + n] = (_Float16)gelu_f(v);
      }
    }
  }
  __syncthreads();

  // stage 2: h2 = gelu(h1 @ W2 + b2)             (K=128, B frags from L2)
  {
    v8f acc[8] = {};
    for (int kt = 0; kt < 4; ++kt) {
      const v16h a = load_a_frag(hbuf, HID, w * 16, kt * 32, lane);
#pragma unroll
      for (int nt = 0; nt < 8; ++nt) {
        const v16h bf = load_b_frag(W2h, HID, kt * 32, nt * 16, lane);
        acc[nt] = WMMA_F16(a, bf, acc[nt]);
      }
    }
    __syncthreads();
#pragma unroll
    for (int nt = 0; nt < 8; ++nt) {
      const int n = nt * 16 + nn;
      const float bb = b2[n];
#pragma unroll
      for (int r = 0; r < 8; ++r) {
        const int m = w * 16 + r + 8 * mh;
        hbuf[m * HID + n] = (_Float16)gelu_f(acc[nt][r] + bb);
      }
    }
  }
  __syncthreads();

  // stage 3: u = h2 @ W3 + b3; pred = (z - u)*scale + loc
  {
    v8f acc[2] = {};
    for (int kt = 0; kt < 4; ++kt) {
      const v16h a = load_a_frag(hbuf, HID, w * 16, kt * 32, lane);
#pragma unroll
      for (int nt = 0; nt < 2; ++nt) {
        const v16h bf = load_b_frag(W3h, 32, kt * 32, nt * 16, lane);
        acc[nt] = WMMA_F16(a, bf, acc[nt]);
      }
    }
#pragma unroll
    for (int nt = 0; nt < 2; ++nt) {
      const int p = nt * 16 + nn;
      if (p < PRED) {
        const float bb = b3[p];
#pragma unroll
        for (int r = 0; r < 8; ++r) {
          const int m = w * 16 + r + 8 * mh;
          const int bg = rowBase + m;
          const float u = acc[nt][r] + bb;
          const float z = z_normal((unsigned)(s * (BATCH * PRED) + bg * PRED + p));
          out[(size_t)bg * (NSAMP * PRED) + s * PRED + p] =
              (z - u) * scalep[bg] + locp[bg];
        }
      }
    }
  }
}

// ---------------------------------------------------------------------------
extern "C" void kernel_launch(void* const* d_in, const int* in_sizes, int n_in,
                              void* d_out, int out_size, void* d_ws, size_t ws_size,
                              hipStream_t stream) {
  const float* past_target      = (const float*)d_in[0];
  // d_in[1] past_observed_values: unused by reference forward
  const float* past_time_feat   = (const float*)d_in[2];
  const float* future_time_feat = (const float*)d_in[3];
  const float* W1 = (const float*)d_in[4];
  const float* b1 = (const float*)d_in[5];
  const float* W2 = (const float*)d_in[6];
  const float* b2 = (const float*)d_in[7];
  const float* W3 = (const float*)d_in[8];
  const float* b3 = (const float*)d_in[9];
  float* out = (float*)d_out;

  // workspace layout (f16 region then f32 region), ~11.7 MB total
  _Float16* Xs   = (_Float16*)d_ws;                 // BATCH x K1
  _Float16* W1s  = Xs  + (size_t)BATCH * K1;        // K1 x HID
  _Float16* W1z  = W1s + (size_t)K1 * HID;          // 32 x HID
  _Float16* W2h  = W1z + 32 * HID;                  // HID x HID
  _Float16* W3h  = W2h + HID * HID;                 // HID x 32
  float*    basep  = (float*)(W3h + HID * 32);      // BATCH x HID
  float*    locp   = basep + (size_t)BATCH * HID;   // BATCH
  float*    scalep = locp + BATCH;                  // BATCH

  const int convN = K1 * HID + 32 * HID + HID * HID + HID * 32;  // 643072
  fc_convert<<<(convN + 255) / 256, 256, 0, stream>>>(W1, W2, W3, W1s, W1z, W2h, W3h);
  fc_features<<<BATCH, 256, 0, stream>>>(past_target, past_time_feat,
                                         future_time_feat, Xs, locp, scalep);
  fc_base_gemm<<<BATCH / 128, 256, 0, stream>>>(Xs, W1s, b1, basep);
  fc_samples<<<dim3(BATCH / 128, NSAMP), 256, 0, stream>>>(
      W1z, W2h, W3h, basep, b2, b3, locp, scalep, out);
}
